// CSFAProV2_78675210928474
// MI455X (gfx1250) — matrix operand order repaired and checked by reading, not compiled, not run on hardware
//
#include <hip/hip_runtime.h>
#include <math.h>

// ---------------------------------------------------------------------------
// CDNA5 (gfx1250) bf16 WMMA types
// ---------------------------------------------------------------------------
typedef __attribute__((ext_vector_type(16))) __bf16        v16bf;
typedef __attribute__((ext_vector_type(8)))  float         v8f;
typedef __attribute__((ext_vector_type(4)))  unsigned int  u32x4;
typedef int v4i_t __attribute__((vector_size(16)));   // matches builtin param type

union Frag32 {            // 32 bytes = 16 bf16 = one WMMA A or B operand per lane
    u32x4 q[2];
    v16bf v;
};

// gfx1250 async global->LDS copy (ASYNCcnt) if the toolchain exposes it.
#if defined(__gfx1250__) && __has_builtin(__builtin_amdgcn_global_load_async_to_lds_b128)
#define USE_ASYNC_COPY 1
#else
#define USE_ASYNC_COPY 0
#endif

#if USE_ASYNC_COPY
__device__ __forceinline__ void async_copy16(const void* g, void* l) {
    __builtin_amdgcn_global_load_async_to_lds_b128(
        (__attribute__((address_space(1))) v4i_t*)(unsigned long long)(uintptr_t)g,
        (__attribute__((address_space(3))) v4i_t*)(unsigned int)(uintptr_t)l,
        0, 0);
}
__device__ __forceinline__ void wait_async0() {
#if __has_builtin(__builtin_amdgcn_s_wait_asynccnt)
    __builtin_amdgcn_s_wait_asynccnt(0);
#else
    asm volatile("s_wait_asynccnt 0x0" ::: "memory");
#endif
}
#endif

__device__ __forceinline__ float swish(float y) {
    return y * (1.0f / (1.0f + __expf(-y)));
}

// ---------------------------------------------------------------------------
// Weight packer: OIHW fp32 -> bf16 in B-fragment order.
// Layout: [t][kc][ntile][lane][j] halves, with B-matrix (32K x 16N) mapping:
//   k = j + ((lane>>4)<<4),  n = ntile*16 + (lane&15)
// ---------------------------------------------------------------------------
__global__ void pack_w_kernel(const float* __restrict__ w, __bf16* __restrict__ wf,
                              int CO, int CI, int KS, size_t total) {
    size_t i = (size_t)blockIdx.x * blockDim.x + threadIdx.x;
    if (i >= total) return;
    const int j   = (int)(i & 15);
    const int lnn = (int)((i >> 4) & 31);
    size_t rest   = i >> 9;
    const int NT  = CO >> 4;
    const int KCH = CI >> 5;
    const int ntile = (int)(rest % NT); rest /= NT;
    const int kc    = (int)(rest % KCH);
    const int t     = (int)(rest / KCH);
    const int kk = j + ((lnn >> 4) << 4);
    const int ci = kc * 32 + kk;
    const int n  = ntile * 16 + (lnn & 15);
    const int ky = t / KS, kx = t - ky * KS;
    wf[i] = (__bf16)w[(((size_t)n * CI + ci) * KS + ky) * KS + kx];
}

// ---------------------------------------------------------------------------
// Implicit-GEMM conv with fused scale/bias/swish (+ optional bf16 residual).
// Block: 128 threads (4 waves, 2x2), tile 64(M) x 64(N).  K = KS*KS*CIN in
// chunks of 32 (bf16 WMMA 16x16x32).  A double-buffered in LDS (async
// global->LDS copies where available), B straight from the fragment-packed
// weight buffer (L2-resident).
// OMODE: 0 -> bf16 NHWC, 1 -> fp32 NCHW into d_out (chan_base), 2 -> fp32 NHWC
// ---------------------------------------------------------------------------
template<int CIN0, int CIN1, int KS, int STRIDE, int IH, int IW, int OH, int OW,
         int COUT, int RESID, int OMODE>
__global__ __launch_bounds__(128)
void conv_wmma(const __bf16* __restrict__ in0, const __bf16* __restrict__ in1,
               const __bf16* __restrict__ wfrag,
               const float* __restrict__ scale, const float* __restrict__ bias,
               const __bf16* __restrict__ resid,
               __bf16* __restrict__ outH, float* __restrict__ outF, int chan_base)
{
    constexpr int CIN  = CIN0 + CIN1;
    constexpr int PAD  = KS >> 1;
    constexpr int KCH  = CIN / 32;       // 32-wide K chunks per tap
    constexpr int NT   = COUT / 16;      // total 16-wide N tiles
    constexpr int ASTR = 40;             // LDS row stride in halves (conflict-free)
    constexpr int NK   = KS * KS * KCH;  // total K chunks

    __shared__ __attribute__((aligned(16))) __bf16 As[2][64 * ASTR];

    const int tid = threadIdx.x;
    const int ln  = tid & 31;
    const int wv  = tid >> 5;
    const int wm  = wv >> 1;             // wave M index (0..1)
    const int wn  = wv & 1;              // wave N index (0..1)
    const int bm0 = blockIdx.x * 64;
    const int bn0 = blockIdx.y * 64;

    v8f acc[2][2] = {};

    // A-staging coordinates: each thread moves 32B (16 halves) per K-chunk
    const int sm  = tid >> 1;            // tile row 0..63
    const int sh  = tid & 1;             // channel half (16 channels)
    const int gm  = bm0 + sm;
    const int sb  = gm / (OH * OW);
    const int srm = gm - sb * (OH * OW);
    const int soy = srm / OW;
    const int sox = srm - soy * OW;

    const int kh = (ln >> 4) << 3;       // A lane-half K offset is 2*kh (0 or 16)

    // global source for chunk q (nullptr => zero-fill, halo)
    auto src_ptr = [&](int q) -> const __bf16* {
        const int t  = q / KCH, kc = q - t * KCH;
        const int ky = t / KS,  kx = t - ky * KS;
        const int iy = soy * STRIDE + ky - PAD;
        const int ix = sox * STRIDE + kx - PAD;
        if (iy < 0 || iy >= IH || ix < 0 || ix >= IW) return nullptr;
        const int ci0 = kc * 32 + sh * 16;
        if (CIN1 == 0 || ci0 < CIN0)
            return in0 + ((size_t)(sb * IH + iy) * IW + ix) * CIN0 + ci0;
        return in1 + ((size_t)(sb * IH + iy) * IW + ix) * CIN1 + (ci0 - CIN0);
    };

#if USE_ASYNC_COPY
    auto stage = [&](int q, int buf) {
        __bf16* dst = &As[buf][sm * ASTR + sh * 16];
        const __bf16* s = src_ptr(q);
        if (s) {
            async_copy16(s,     dst);          // ASYNCcnt-tracked DMA to LDS
            async_copy16(s + 8, dst + 8);
        } else {
            const u32x4 z = {0, 0, 0, 0};
            reinterpret_cast<u32x4*>(dst)[0] = z;
            reinterpret_cast<u32x4*>(dst)[1] = z;
        }
    };
    stage(0, 0);
    wait_async0();
    __syncthreads();
#else
    {
        u32x4 r0 = {0, 0, 0, 0}, r1 = {0, 0, 0, 0};
        const __bf16* s = src_ptr(0);
        if (s) {
            const u32x4* g = reinterpret_cast<const u32x4*>(s);
            r0 = g[0]; r1 = g[1];
        }
        u32x4* d = reinterpret_cast<u32x4*>(&As[0][sm * ASTR + sh * 16]);
        d[0] = r0; d[1] = r1;
    }
    __syncthreads();
#endif

    for (int q = 0; q < NK; ++q) {
        const int cur = q & 1;

        // prefetch chunk q+1 into the other buffer while computing chunk q
#if USE_ASYNC_COPY
        if (q + 1 < NK) stage(q + 1, cur ^ 1);
#else
        if (q + 1 < NK) {
            u32x4 r0 = {0, 0, 0, 0}, r1 = {0, 0, 0, 0};
            const __bf16* s = src_ptr(q + 1);
            if (s) {
                const u32x4* g = reinterpret_cast<const u32x4*>(s);
                r0 = g[0]; r1 = g[1];
            }
            u32x4* d = reinterpret_cast<u32x4*>(&As[cur ^ 1][sm * ASTR + sh * 16]);
            d[0] = r0; d[1] = r1;
        }
#endif
        // B fragments: contiguous 32B per lane (weight index == chunk index q)
        Frag32 bfr[2];
        #pragma unroll
        for (int nf = 0; nf < 2; ++nf) {
            const int ntile = (bn0 >> 4) + wn * 2 + nf;
            const u32x4* wp = reinterpret_cast<const u32x4*>(
                wfrag + (((size_t)q * NT + ntile) << 9) + (ln << 4));
            bfr[nf].q[0] = wp[0];
            bfr[nf].q[1] = wp[1];
            __builtin_prefetch(wp + 2 * NT, 0, 1);   // 32*NT bytes ahead in the weights
        }
        // A fragments: 16-bit A 16x32 layout -> two 16B LDS reads per lane
        Frag32 afr[2];
        #pragma unroll
        for (int mf = 0; mf < 2; ++mf) {
            const int m = wm * 32 + mf * 16 + (ln & 15);
            afr[mf].q[0] = *reinterpret_cast<const u32x4*>(&As[cur][m * ASTR + 2 * kh]);
            afr[mf].q[1] = *reinterpret_cast<const u32x4*>(&As[cur][m * ASTR + 2 * kh + 8]);
        }
        #pragma unroll
        for (int mf = 0; mf < 2; ++mf)
            #pragma unroll
            for (int nf = 0; nf < 2; ++nf)
                acc[mf][nf] = __builtin_amdgcn_wmma_f32_16x16x32_bf16(
                    false, afr[mf].v, false, bfr[nf].v,
                    (short)0, acc[mf][nf], false, false);

#if USE_ASYNC_COPY
        if (q + 1 < NK) wait_async0();   // after the math: overlap DMA with WMMA
#endif
        __syncthreads();
    }

    // Epilogue: y = acc*scale+bias ; swish ; (+resid) ; store
    #pragma unroll
    for (int mf = 0; mf < 2; ++mf) {
        #pragma unroll
        for (int nf = 0; nf < 2; ++nf) {
            const int n  = bn0 + wn * 32 + nf * 16 + (ln & 15);
            const float sc = scale[n];
            const float bi = bias[n];
            #pragma unroll
            for (int r = 0; r < 8; ++r) {
                const int m   = bm0 + wm * 32 + mf * 16 + r + ((ln >> 4) << 3);
                const int b   = m / (OH * OW);
                const int rem = m - b * (OH * OW);
                const int oy  = rem / OW;
                const int ox  = rem - oy * OW;
                float y = acc[mf][nf][r] * sc + bi;
                float o = swish(y);
                const size_t pix = (size_t)(b * OH + oy) * OW + ox;
                if constexpr (RESID != 0) o += (float)resid[pix * COUT + n];
                if constexpr (OMODE == 0)
                    outH[pix * COUT + n] = (__bf16)o;
                else if constexpr (OMODE == 1)
                    outF[((size_t)b * 1024 + chan_base + n) * (OH * OW) + rem] = o;
                else
                    outF[pix * COUT + n] = o;
            }
        }
    }
}

// ---------------------------------------------------------------------------
// Elementwise / packing helpers
// ---------------------------------------------------------------------------
__global__ void pack_x1h_kernel(const float* __restrict__ x1, __bf16* __restrict__ x1h) {
    size_t i = (size_t)blockIdx.x * blockDim.x + threadIdx.x;
    if (i >= (size_t)32 * 20 * 20 * 512) return;
    const int c = (int)(i & 511);
    const int x = (int)((i >> 9) % 20);
    const int y = (int)((i >> 9) / 20 % 20);
    const int b = (int)(i >> 9) / 400;
    x1h[i] = (__bf16)x1[(((size_t)b * 512 + c) * 20 + y) * 20 + x];
}

// Upsampled x1 (both halves), NHWC bf16 at 40x40: feeds conv2 channels [0,512)
__global__ void pack_a01_kernel(const float* __restrict__ x1, __bf16* __restrict__ a01) {
    size_t i = (size_t)blockIdx.x * blockDim.x + threadIdx.x;
    if (i >= (size_t)32 * 40 * 40 * 512) return;
    const int c = (int)(i & 511);
    const int x = (int)((i >> 9) % 40);
    const int y = (int)((i >> 9) / 40 % 40);
    const int b = (int)(i >> 9) / 1600;
    a01[i] = (__bf16)x1[(((size_t)b * 512 + c) * 20 + (y >> 1)) * 20 + (x >> 1)];
}

// kp[b][l][c] = mean over 4x4 patch of x2
__global__ void kp_pool_kernel(const float* __restrict__ x2, float* __restrict__ kp) {
    size_t i = (size_t)blockIdx.x * blockDim.x + threadIdx.x;
    if (i >= (size_t)32 * 100 * 512) return;
    const int c = (int)(i & 511);
    const int l = (int)((i >> 9) % 100);
    const int b = (int)((i >> 9) / 100);
    const int ph = l / 10, pw = l - ph * 10;
    float s = 0.f;
    const float* base = x2 + (((size_t)b * 512 + c) * 40 + ph * 4) * 40 + pw * 4;
    #pragma unroll
    for (int py = 0; py < 4; ++py)
        #pragma unroll
        for (int px = 0; px < 4; ++px)
            s += base[py * 40 + px];
    kp[i] = s * 0.0625f;
}

// k[b][l][e] = sum_c key_w[e][c] * kp[b][l][c]
__global__ void kproj_kernel(const float* __restrict__ kp, const float* __restrict__ kw,
                             float* __restrict__ kmat) {
    size_t i = (size_t)blockIdx.x * blockDim.x + threadIdx.x;
    if (i >= (size_t)32 * 100 * 256) return;
    const int e  = (int)(i & 255);
    const size_t bl = i >> 8;
    const float* a = kp + bl * 512;
    const float* w = kw + (size_t)e * 512;
    float s = 0.f;
    for (int c = 0; c < 512; ++c) s += a[c] * w[c];
    kmat[i] = s;
}

// score + softmax per (b*NH, q-row); writes probs straight into WMMA
// A-fragment layout [bnh][mf(7)][kc(4)][lane(32)][j(16)], K padded to 128.
__global__ __launch_bounds__(128)
void score_softmax_kernel(const float* __restrict__ q, const float* __restrict__ k,
                          __bf16* __restrict__ wtsA) {
    const int blk  = blockIdx.x;
    const int bnh  = blk / 100;
    const int qrow = blk - bnh * 100;
    const int b = bnh >> 3, nh = bnh & 7;
    __shared__ float sc[128];
    __shared__ float red[128];
    const int tid = threadIdx.x;
    float v = -1e30f;
    if (tid < 100) {
        const float* qp = q + ((size_t)b * 100 + qrow) * 256 + nh * 32;
        const float* kp = k + ((size_t)b * 100 + tid)  * 256 + nh * 32;
        float s = 0.f;
        #pragma unroll
        for (int d = 0; d < 32; ++d) s += qp[d] * kp[d];
        v = s * 0.17677669529663689f;               // 1/sqrt(32)
    }
    sc[tid] = v; red[tid] = v; __syncthreads();
    for (int o = 64; o > 0; o >>= 1) {
        if (tid < o) red[tid] = fmaxf(red[tid], red[tid + o]);
        __syncthreads();
    }
    const float mx = red[0]; __syncthreads();
    const float e = (tid < 100) ? __expf(sc[tid] - mx) : 0.f;
    red[tid] = e; __syncthreads();
    for (int o = 64; o > 0; o >>= 1) {
        if (tid < o) red[tid] += red[tid + o];
        __syncthreads();
    }
    const float w = e / red[0];
    // A-matrix 16x32 mapping: k = (j&7) + ((j>>3)<<4) + ((lane>>4)<<3)
    const int mf   = qrow >> 4, lnlo = qrow & 15;
    const int kc   = tid >> 5,  kk   = tid & 31;
    const int jj   = (kk & 7) | (((kk >> 4) & 1) << 3);
    const int lane = lnlo + (((kk >> 3) & 1) << 4);
    wtsA[((((size_t)bnh * 7 + mf) * 4 + kc) * 32 + lane) * 16 + jj] = (__bf16)w;
}

// vT[bnh][col][l] bf16, l padded 100->128 with zeros (B operand of y-GEMM)
__global__ void pack_v_kernel(const float* __restrict__ x2, __bf16* __restrict__ vT) {
    size_t i = (size_t)blockIdx.x * blockDim.x + threadIdx.x;
    if (i >= (size_t)256 * 1024 * 128) return;
    const int l   = (int)(i & 127);
    const int col = (int)((i >> 7) & 1023);
    const int bnh = (int)(i >> 17);
    float val = 0.f;
    if (l < 100) {
        const int b = bnh >> 3, nh = bnh & 7;
        const int chh = col >> 4, py = (col >> 2) & 3, px = col & 3;
        const int ph = l / 10, pw = l - ph * 10;
        const int c = nh * 64 + chh, h = ph * 4 + py, w = pw * 4 + px;
        val = x2[(((size_t)b * 512 + c) * 40 + h) * 40 + w];
    }
    vT[i] = (__bf16)val;
}

// y = wts @ v, batched over 256 heads.  No LDS: A from fragment-packed wtsA,
// B from transposed vT (A: 32B contiguous per lane; B: two 16B reads at +0B, +32B).  Scatters into bf16
// NHWC attn buffer.
__global__ __launch_bounds__(256)
void attn_y_kernel(const __bf16* __restrict__ wtsA, const __bf16* __restrict__ vT,
                   __bf16* __restrict__ attnH) {
    const int bnh = blockIdx.y;                 // 256 batches
    const int nb  = blockIdx.x;                 // 8 x 128 cols
    const int tid = threadIdx.x;
    const int ln  = tid & 31;
    const int wv  = tid >> 5;                   // 8 waves, 16 cols each
    const int n0  = nb * 128 + wv * 16;

    v8f acc[7] = {};
    #pragma unroll
    for (int kc = 0; kc < 4; ++kc) {
        Frag32 bfr;
        const u32x4* vp = reinterpret_cast<const u32x4*>(
            vT + (((size_t)bnh * 1024 + n0 + (ln & 15)) << 7) + kc * 32 + ((ln >> 4) << 3));
        bfr.q[0] = vp[0]; bfr.q[1] = vp[2];
        #pragma unroll
        for (int mf = 0; mf < 7; ++mf) {
            Frag32 afr;
            const u32x4* ap = reinterpret_cast<const u32x4*>(
                wtsA + ((((size_t)bnh * 7 + mf) * 4 + kc) << 9) + (ln << 4));
            afr.q[0] = ap[0]; afr.q[1] = ap[1];
            acc[mf] = __builtin_amdgcn_wmma_f32_16x16x32_bf16(
                false, afr.v, false, bfr.v, (short)0, acc[mf], false, false);
        }
    }
    const int b = bnh >> 3, nh = bnh & 7;
    const int n = n0 + (ln & 15);
    const int chh = n >> 4, py = (n >> 2) & 3, px = n & 3;
    #pragma unroll
    for (int mf = 0; mf < 7; ++mf) {
        #pragma unroll
        for (int r = 0; r < 8; ++r) {
            const int m = mf * 16 + r + ((ln >> 4) << 3);
            if (m < 100) {
                const int ph = m / 10, pw = m - ph * 10;
                const int c = nh * 64 + chh, h = ph * 4 + py, w = pw * 4 + px;
                attnH[(((size_t)b * 40 + h) * 40 + w) * 512 + c] = (__bf16)acc[mf][r];
            }
        }
    }
}

// channel-attention MLP: g = sigmoid(W2 relu(W1 p + b1) + b2), per 20x20 pixel
__global__ __launch_bounds__(256)
void chan_att_kernel(const float* __restrict__ x1,
                     const float* __restrict__ w1, const float* __restrict__ b1,
                     const float* __restrict__ w2, const float* __restrict__ b2,
                     float* __restrict__ g) {
    const int pix = blockIdx.x;                 // b*400 + y0*20 + x0
    const int b   = pix / 400;
    const int rem = pix - b * 400;
    const int y0  = rem / 20, x0 = rem - y0 * 20;
    __shared__ float p[256];
    __shared__ float hdn[64];
    const int tid = threadIdx.x;
    p[tid] = x1[(((size_t)b * 512 + 256 + tid) * 20 + y0) * 20 + x0];
    __syncthreads();
    if (tid < 64) {
        float s = b1[tid];
        const float* wr = w1 + (size_t)tid * 256;
        for (int c = 0; c < 256; ++c) s += wr[c] * p[c];
        hdn[tid] = fmaxf(s, 0.f);
    }
    __syncthreads();
    float s = b2[tid];
    const float* wr = w2 + (size_t)tid * 64;
    #pragma unroll
    for (int j = 0; j < 64; ++j) s += wr[j] * hdn[j];
    g[(size_t)pix * 256 + tid] = 1.f / (1.f + __expf(-s));
}

// x_ca = a1 * g (broadcast g from 20x20 to 40x40), bf16 NHWC
__global__ void make_xca_kernel(const float* __restrict__ x1, const float* __restrict__ g,
                                __bf16* __restrict__ xca) {
    size_t i = (size_t)blockIdx.x * blockDim.x + threadIdx.x;
    if (i >= (size_t)32 * 40 * 40 * 256) return;
    const int c = (int)(i & 255);
    const int x = (int)((i >> 8) % 40);
    const int y = (int)((i >> 8) / 40 % 40);
    const int b = (int)(i >> 8) / 1600;
    const int y0 = y >> 1, x0 = x >> 1;
    const float a1 = x1[(((size_t)b * 512 + 256 + c) * 20 + y0) * 20 + x0];
    const float gg = g[((size_t)(b * 20 + y0) * 20 + x0) * 256 + c];
    xca[i] = (__bf16)(a1 * gg);
}

// ---------------------------------------------------------------------------
// Launch
// ---------------------------------------------------------------------------
static inline unsigned cdiv(size_t a, unsigned b) { return (unsigned)((a + b - 1) / b); }

extern "C" void kernel_launch(void* const* d_in, const int* in_sizes, int n_in,
                              void* d_out, int out_size, void* d_ws, size_t ws_size,
                              hipStream_t stream) {
    (void)in_sizes; (void)n_in; (void)out_size; (void)ws_size;
    const float* x1   = (const float*)d_in[0];
    const float* x2   = (const float*)d_in[1];
    const float* q_w  = (const float*)d_in[2];
    const float* q_s  = (const float*)d_in[3];
    const float* q_b  = (const float*)d_in[4];
    const float* keyw = (const float*)d_in[5];
    const float* caw1 = (const float*)d_in[6];
    const float* cab1 = (const float*)d_in[7];
    const float* caw2 = (const float*)d_in[8];
    const float* cab2 = (const float*)d_in[9];
    const float* sa1w = (const float*)d_in[10];
    const float* sa1s = (const float*)d_in[11];
    const float* sa1b = (const float*)d_in[12];
    const float* sa2w = (const float*)d_in[13];
    const float* sa2s = (const float*)d_in[14];
    const float* sa2b = (const float*)d_in[15];
    const float* c2w  = (const float*)d_in[16];
    const float* c2s  = (const float*)d_in[17];
    const float* c2b  = (const float*)d_in[18];
    const float* c3w  = (const float*)d_in[19];
    const float* c3s  = (const float*)d_in[20];
    const float* c3b  = (const float*)d_in[21];
    float* out = (float*)d_out;

    char* ws = (char*)d_ws;
    size_t off = 0;
    auto alloc = [&](size_t bytes) -> char* {
        char* p = ws + off;
        off += (bytes + 255) & ~(size_t)255;
        return p;
    };
    __bf16* x1h   = (__bf16*)alloc((size_t)32 * 20 * 20 * 512 * 2);
    __bf16* a01   = (__bf16*)alloc((size_t)32 * 40 * 40 * 512 * 2);
    __bf16* Wq    = (__bf16*)alloc((size_t)9 * 512 * 256 * 2);
    __bf16* Wc3   = (__bf16*)alloc((size_t)9 * 512 * 512 * 2);
    __bf16* Wsa1  = (__bf16*)alloc((size_t)9 * 256 * 256 * 2);
    __bf16* Wsa2  = (__bf16*)alloc((size_t)9 * 256 * 256 * 2);
    __bf16* Wc2   = (__bf16*)alloc((size_t)768 * 512 * 2);
    float*  kpbuf = (float*)alloc((size_t)32 * 100 * 512 * 4);
    float*  kmat  = (float*)alloc((size_t)32 * 100 * 256 * 4);
    float*  qmat  = (float*)alloc((size_t)32 * 100 * 256 * 4);
    __bf16* wtsA  = (__bf16*)alloc((size_t)256 * 7 * 4 * 32 * 16 * 2);
    __bf16* vT    = (__bf16*)alloc((size_t)256 * 1024 * 128 * 2);
    __bf16* attnH = (__bf16*)alloc((size_t)32 * 40 * 40 * 512 * 2);
    float*  gbuf  = (float*)alloc((size_t)32 * 20 * 20 * 256 * 4);
    __bf16* xcaH  = (__bf16*)alloc((size_t)32 * 40 * 40 * 256 * 2);
    __bf16* t1H   = (__bf16*)alloc((size_t)32 * 40 * 40 * 256 * 2);
    __bf16* a2H   = (__bf16*)alloc((size_t)32 * 40 * 40 * 256 * 2);

    // ---- weight packing (fragment order) ----
    {
        size_t t;
        t = (size_t)9 * 512 * 256;
        pack_w_kernel<<<cdiv(t, 256), 256, 0, stream>>>(q_w,  Wq,  256, 512, 3, t);
        t = (size_t)9 * 512 * 512;
        pack_w_kernel<<<cdiv(t, 256), 256, 0, stream>>>(c3w,  Wc3, 512, 512, 3, t);
        t = (size_t)9 * 256 * 256;
        pack_w_kernel<<<cdiv(t, 256), 256, 0, stream>>>(sa1w, Wsa1, 256, 256, 3, t);
        pack_w_kernel<<<cdiv(t, 256), 256, 0, stream>>>(sa2w, Wsa2, 256, 256, 3, t);
        t = (size_t)768 * 512;
        pack_w_kernel<<<cdiv(t, 256), 256, 0, stream>>>(c2w,  Wc2, 512, 768, 1, t);
    }
    // ---- activation packing ----
    pack_x1h_kernel<<<cdiv((size_t)32 * 400 * 512, 256), 256, 0, stream>>>(x1, x1h);
    pack_a01_kernel<<<cdiv((size_t)32 * 1600 * 512, 256), 256, 0, stream>>>(x1, a01);

    // ---- attention: keys ----
    kp_pool_kernel<<<cdiv((size_t)32 * 100 * 512, 256), 256, 0, stream>>>(x2, kpbuf);
    kproj_kernel<<<cdiv((size_t)32 * 100 * 256, 256), 256, 0, stream>>>(kpbuf, keyw, kmat);

    // ---- attention: queries = WMMA conv3x3 stride2 on x1 ----
    conv_wmma<512, 0, 3, 2, 20, 20, 10, 10, 256, 0, 2>
        <<<dim3(50, 4), 128, 0, stream>>>(x1h, x1h, Wq, q_s, q_b, nullptr,
                                          nullptr, qmat, 0);

    // ---- score + softmax (A-fragment layout) ----
    score_softmax_kernel<<<256 * 100, 128, 0, stream>>>(qmat, kmat, wtsA);

    // ---- v pack (transposed, zero-padded K) ----
    pack_v_kernel<<<cdiv((size_t)256 * 1024 * 128, 256), 256, 0, stream>>>(x2, vT);

    // ---- y = wts @ v  (batched WMMA GEMM, no LDS) ----
    attn_y_kernel<<<dim3(8, 256), 256, 0, stream>>>(wtsA, vT, attnH);

    // ---- x2_out = swish(conv3x3(attn)*s+b) + attn -> out channels [512,1024) ----
    conv_wmma<512, 0, 3, 1, 40, 40, 40, 40, 512, 1, 1>
        <<<dim3(800, 8), 128, 0, stream>>>(attnH, attnH, Wc3, c3s, c3b, attnH,
                                           nullptr, out, 512);

    // ---- x1 branch: channel attention, spatial convs ----
    chan_att_kernel<<<32 * 400, 256, 0, stream>>>(x1, caw1, cab1, caw2, cab2, gbuf);
    make_xca_kernel<<<cdiv((size_t)32 * 1600 * 256, 256), 256, 0, stream>>>(x1, gbuf, xcaH);

    conv_wmma<256, 0, 3, 1, 40, 40, 40, 40, 256, 0, 0>
        <<<dim3(800, 4), 128, 0, stream>>>(xcaH, xcaH, Wsa1, sa1s, sa1b, nullptr,
                                           t1H, nullptr, 0);
    conv_wmma<256, 0, 3, 1, 40, 40, 40, 40, 256, 1, 0>
        <<<dim3(800, 4), 128, 0, stream>>>(t1H, t1H, Wsa2, sa2s, sa2b, xcaH,
                                           a2H, nullptr, 0);

    // ---- x1_out = swish(1x1 conv over cat[a0,a1,a2]) -> out channels [0,512) ----
    conv_wmma<512, 256, 1, 1, 40, 40, 40, 40, 512, 0, 1>
        <<<dim3(800, 8), 128, 0, stream>>>(a01, a2H, Wc2, c2s, c2b, nullptr,
                                           nullptr, out, 0);
}
